// TinyAttention_62843961475101
// MI455X (gfx1250) — compile-verified, matrix operand order checked by reading, not run on hardware
//
#include <hip/hip_runtime.h>
#include <hip/hip_bf16.h>

#define TT 8192
#define CC 512
#define AA 64

typedef __attribute__((ext_vector_type(16))) __bf16 v16bf;
typedef __attribute__((ext_vector_type(8)))  float  v8f;

union Frag16 {
  v16bf v;
  uint4 q[2];
  unsigned int u[8];
};

__device__ __forceinline__ unsigned short f32_to_bf16(float f) {
  unsigned int u = __float_as_uint(f);
  u += 0x7fffu + ((u >> 16) & 1u);   // round-to-nearest-even
  return (unsigned short)(u >> 16);
}

__device__ __forceinline__ unsigned int pack_bf16(float lo, float hi) {
  return (unsigned int)f32_to_bf16(lo) | ((unsigned int)f32_to_bf16(hi) << 16);
}

__device__ __forceinline__ v8f wmma_bf16(const Frag16& a, const Frag16& b, v8f c) {
  // (neg_a, A, neg_b, B, c_mod, C, reuse_a, reuse_b)
  return __builtin_amdgcn_wmma_f32_16x16x32_bf16(false, a.v, false, b.v,
                                                 (short)0, c, false, false);
}

__device__ __forceinline__ v8f v8f_zero() {
  v8f z = {0.f, 0.f, 0.f, 0.f, 0.f, 0.f, 0.f, 0.f};
  return z;
}

// Async global->LDS copy of 16 bytes per lane (tracked by ASYNCcnt).
__device__ __forceinline__ void async_load_b128(unsigned lds_off,
                                                const void* gptr) {
  asm volatile("global_load_async_to_lds_b128 %0, %1, off"
               :: "v"(lds_off), "v"((unsigned long long)(size_t)gptr)
               : "memory");
}

// ---------------------------------------------------------------------------
// Kernel 0a: WvT[c][r] = bf16(Wv[r][c])   (512x512)
// ---------------------------------------------------------------------------
__global__ __launch_bounds__(256) void wv_transpose_kernel(
    const float* __restrict__ Wv, unsigned short* __restrict__ wvt) {
  const int idx = blockIdx.x * 256 + threadIdx.x;   // 0 .. 512*512-1
  const int r = idx >> 9;
  const int c = idx & 511;
  wvt[c * CC + r] = f32_to_bf16(Wv[idx]);
}

// ---------------------------------------------------------------------------
// Kernel 0b: xembB = bf16(x_emb), row-major, 8 elements per thread
// ---------------------------------------------------------------------------
__global__ __launch_bounds__(256) void xemb_cvt_kernel(
    const float* __restrict__ xemb, unsigned short* __restrict__ xembB) {
  const size_t i8 = (size_t)(blockIdx.x * 256 + threadIdx.x) * 8;
  const float4 f0 = *(const float4*)(xemb + i8);
  const float4 f1 = *(const float4*)(xemb + i8 + 4);
  uint4 st;
  st.x = pack_bf16(f0.x, f0.y);
  st.y = pack_bf16(f0.z, f0.w);
  st.z = pack_bf16(f1.x, f1.y);
  st.w = pack_bf16(f1.z, f1.w);
  *(uint4*)(xembB + i8) = st;
}

// ---------------------------------------------------------------------------
// Kernel 1: one row per block. LayerNorm(x[t]) then q = xx@Wq * 0.125 (bf16),
// k = xx@Wk (bf16).
// ---------------------------------------------------------------------------
__global__ __launch_bounds__(256) void ln_qk_kernel(
    const float* __restrict__ x, const float* __restrict__ sc,
    const float* __restrict__ bi, const float* __restrict__ Wq,
    const float* __restrict__ Wk, unsigned short* __restrict__ qb,
    unsigned short* __restrict__ kb) {
  __shared__ float xs[CC];
  __shared__ float red[256];
  const int tid = threadIdx.x;
  const size_t t = blockIdx.x;

  const float v0 = x[t * CC + tid];
  const float v1 = x[t * CC + 256 + tid];

  red[tid] = v0 + v1;
  __syncthreads();
  for (int off = 128; off > 0; off >>= 1) {
    if (tid < off) red[tid] += red[tid + off];
    __syncthreads();
  }
  const float mu = red[0] * (1.0f / CC);
  __syncthreads();

  const float d0 = v0 - mu, d1 = v1 - mu;
  red[tid] = d0 * d0 + d1 * d1;
  __syncthreads();
  for (int off = 128; off > 0; off >>= 1) {
    if (tid < off) red[tid] += red[tid + off];
    __syncthreads();
  }
  const float rs = rsqrtf(red[0] * (1.0f / CC) + 1e-5f);

  xs[tid]       = d0 * rs * sc[tid] + bi[tid];
  xs[tid + 256] = d1 * rs * sc[tid + 256] + bi[tid + 256];
  __syncthreads();

  if (tid < 128) {
    const int j = tid & 63;
    const float* __restrict__ W = (tid < 64) ? Wq : Wk;
    float acc = 0.0f;
#pragma unroll 8
    for (int kk = 0; kk < CC; ++kk) acc = fmaf(xs[kk], W[kk * AA + j], acc);
    if (tid < 64)
      qb[t * AA + j] = f32_to_bf16(acc * 0.125f);   // fold A^-0.5 into q
    else
      kb[t * AA + j] = f32_to_bf16(acc);
  }
}

// ---------------------------------------------------------------------------
// Kernel 2: v = x_emb @ Wv, stored TRANSPOSED as Vt[c][t] in bf16.
// Wave tile: 16 rows x 64 cols (4 x v8f accumulators), K loop of 32.
// Block = 8 waves -> 128 rows x 64 cols. Grid (T/128, C/64).
// ---------------------------------------------------------------------------
__global__ __launch_bounds__(256) void gemm_v_kernel(
    const unsigned short* __restrict__ xembB,
    const unsigned short* __restrict__ wvt,
    unsigned short* __restrict__ vt) {
  const int w    = threadIdx.x >> 5;
  const int lane = threadIdx.x & 31;
  const int ln   = lane & 15;
  const int hi   = lane >> 4;
  const int rowbase = blockIdx.x * 128 + w * 16;
  const int colbase = blockIdx.y * 64;

  v8f acc[4];
#pragma unroll
  for (int ct = 0; ct < 4; ++ct) acc[ct] = v8f_zero();

  const unsigned short* __restrict__ arow = xembB + (size_t)(rowbase + ln) * CC;

  for (int kb = 0; kb < CC; kb += 32) {
    Frag16 a;   // A: 16x32 bf16, lane ln/hi -> K hi*8+{0..7} and 16+hi*8+{0..7}
    a.q[0] = *(const uint4*)(arow + kb + hi * 8);
    a.q[1] = *(const uint4*)(arow + kb + 16 + hi * 8);
#pragma unroll
    for (int ct = 0; ct < 4; ++ct) {
      // B fragment: column (colbase+ct*16+ln) of Wv == row of WvT, contiguous
      Frag16 b;
      const unsigned short* p =
          wvt + (size_t)(colbase + ct * 16 + ln) * CC + kb + hi * 8;
      b.q[0] = *(const uint4*)p;
      b.q[1] = *(const uint4*)(p + 16);
      acc[ct] = wmma_bf16(a, b, acc[ct]);
    }
  }

  // Store transposed: lane has fixed column n = colbase+ct*16+ln, rows
  // M = hi*8 + i -> consecutive t indices in Vt's row => one b128 store.
#pragma unroll
  for (int ct = 0; ct < 4; ++ct) {
    uint4 st;
    st.x = pack_bf16(acc[ct][0], acc[ct][1]);
    st.y = pack_bf16(acc[ct][2], acc[ct][3]);
    st.z = pack_bf16(acc[ct][4], acc[ct][5]);
    st.w = pack_bf16(acc[ct][6], acc[ct][7]);
    *(uint4*)&vt[(size_t)(colbase + ct * 16 + ln) * TT + rowbase + hi * 8] = st;
  }
}

// ---------------------------------------------------------------------------
// Kernel 3: out = x + tril(q k^T) @ v  (flash-style, no T x T materialization)
// Block = 8 waves = 128 rows x 128 cols; each wave: 16 rows x 128 cols.
// All waves iterate the block-wide causal s-range so the 32x128 V slab can be
// staged cooperatively into LDS with async global->LDS copies (double
// buffered, ASYNCcnt + barriers). Waves whose rows end earlier skip compute
// behind a wave-uniform guard (EXEC stays all-ones around WMMA).
// ---------------------------------------------------------------------------
__global__ __launch_bounds__(256) void attn_pv_kernel(
    const float* __restrict__ x, const unsigned short* __restrict__ qb,
    const unsigned short* __restrict__ kb, const unsigned short* __restrict__ vt,
    float* __restrict__ out) {
  __shared__ unsigned short vslab[2][128][32];  // [buf][col][s] bf16, 2x8KB
  __shared__ unsigned short sbuf[8][16 * 32];   // per-wave 16x32 bf16 S tile

  const int tid  = threadIdx.x;
  const int w    = tid >> 5;
  const int lane = tid & 31;
  const int ln   = lane & 15;
  const int hi   = lane >> 4;
  const int rowbase = blockIdx.x * 128 + w * 16;
  const int colbase = blockIdx.y * 128;

  // cooperative slab copy: thread tid moves 2 x b128 for column (tid>>1)
  const int cpcol  = tid >> 1;                  // 0..127
  const int cppart = (tid & 1) * 2;             // element offset /8: 0 or 2
  const unsigned short* vcol =
      vt + (size_t)(colbase + cpcol) * TT;      // global row for this column

  // q A-fragments: rows rowbase..+15, K = 64 -> two 16x32 fragments
  Frag16 aq[2];
  {
    const unsigned short* qrow = qb + (size_t)(rowbase + ln) * AA;
#pragma unroll
    for (int ks = 0; ks < 2; ++ks) {
      const unsigned short* p = qrow + ks * 32 + hi * 8;
      aq[ks].q[0] = *(const uint4*)p;
      aq[ks].q[1] = *(const uint4*)(p + 16);
    }
  }

  v8f acc[8];
#pragma unroll
  for (int ct = 0; ct < 8; ++ct) acc[ct] = v8f_zero();

  const int nsb = blockIdx.x * 4 + 4;           // block-wide causal s-blocks

  // prologue: start async fill of slab 0
#pragma unroll
  for (int p = 0; p < 2; ++p) {
    async_load_b128((unsigned)(size_t)&vslab[0][cpcol][(cppart + p) * 8],
                    vcol + (cppart + p) * 8);
  }

  for (int sb = 0; sb < nsb; ++sb) {
    const int sbase = sb * 32;
    const int cur = sb & 1;

    if (sb + 1 < nsb) {
      // prefetch next slab into the other buffer, then drain current slab
      const int nxt = cur ^ 1;
      const int nbase = sbase + 32;
#pragma unroll
      for (int p = 0; p < 2; ++p) {
        async_load_b128((unsigned)(size_t)&vslab[nxt][cpcol][(cppart + p) * 8],
                        vcol + nbase + (cppart + p) * 8);
      }
      asm volatile("s_wait_asynccnt 0x2" ::: "memory");
    } else {
      asm volatile("s_wait_asynccnt 0x0" ::: "memory");
    }
    __syncthreads();   // slab `cur` now valid for every wave

    if (sbase <= rowbase + 15) {   // wave-uniform causal guard
      // ---- S = q . k^T  (16 x 32, two 16x16 n-tiles) ----
      v8f s0 = v8f_zero(), s1 = v8f_zero();
#pragma unroll
      for (int ks = 0; ks < 2; ++ks) {
        Frag16 b0, b1;  // B = k^T: column n of B == row (sbase+n) of k
        const unsigned short* p0 =
            kb + (size_t)(sbase + ln) * AA + ks * 32 + hi * 8;
        const unsigned short* p1 = p0 + 16 * AA;
        b0.q[0] = *(const uint4*)p0;  b0.q[1] = *(const uint4*)(p0 + 16);
        b1.q[0] = *(const uint4*)p1;  b1.q[1] = *(const uint4*)(p1 + 16);
        s0 = wmma_bf16(aq[ks], b0, s0);
        s1 = wmma_bf16(aq[ks], b1, s1);
      }

      // ---- causal mask on diagonal-overlapping blocks ----
      if (sbase + 31 > rowbase) {
        const int trow = rowbase + hi * 8;      // C layout: M = hi*8 + i
#pragma unroll
        for (int i = 0; i < 8; ++i) {
          if (sbase + ln > trow + i)      s0[i] = 0.0f;
          if (sbase + 16 + ln > trow + i) s1[i] = 0.0f;
        }
      }

      // ---- C-layout f32 -> A-layout bf16 via wave-private LDS tile ----
#pragma unroll
      for (int i = 0; i < 8; ++i) {
        sbuf[w][(hi * 8 + i) * 32 + ln]      = f32_to_bf16(s0[i]);
        sbuf[w][(hi * 8 + i) * 32 + 16 + ln] = f32_to_bf16(s1[i]);
      }
      asm volatile("s_wait_dscnt 0x0" ::: "memory");  // same-wave DS RAW fence
      Frag16 sa;
      sa.q[0] = *(const uint4*)&sbuf[w][ln * 32 + hi * 8];
      sa.q[1] = *(const uint4*)&sbuf[w][ln * 32 + 16 + hi * 8];

      // ---- O += S (16x32) . V (32x128), V fragments from the LDS slab ----
#pragma unroll
      for (int ct = 0; ct < 8; ++ct) {
        Frag16 bv;
        const unsigned short* p = &vslab[cur][ct * 16 + ln][hi * 8];
        bv.q[0] = *(const uint4*)p;
        bv.q[1] = *(const uint4*)(p + 16);
        acc[ct] = wmma_bf16(sa, bv, acc[ct]);
      }
    }

    __syncthreads();   // everyone done reading slab `cur` before it's refilled
  }

  // ---- epilogue: out = x + acc ----
#pragma unroll
  for (int ct = 0; ct < 8; ++ct) {
    const int c = colbase + ct * 16 + ln;
#pragma unroll
    for (int i = 0; i < 8; ++i) {
      const size_t t = (size_t)(rowbase + hi * 8 + i);
      out[t * CC + c] = x[t * CC + c] + acc[ct][i];
    }
  }
}

// ---------------------------------------------------------------------------
extern "C" void kernel_launch(void* const* d_in, const int* in_sizes, int n_in,
                              void* d_out, int out_size, void* d_ws,
                              size_t ws_size, hipStream_t stream) {
  (void)in_sizes; (void)n_in; (void)out_size; (void)ws_size;
  const float* x    = (const float*)d_in[0];
  const float* xemb = (const float*)d_in[1];
  const float* sc   = (const float*)d_in[2];
  const float* bi   = (const float*)d_in[3];
  const float* Wq   = (const float*)d_in[4];
  const float* Wk   = (const float*)d_in[5];
  const float* Wv   = (const float*)d_in[6];
  float* out = (float*)d_out;

  // ws layout (bf16): q[T*A] | k[T*A] | WvT[C*C] | Vt[C*T] | xembB[T*C]
  unsigned short* qb    = (unsigned short*)d_ws;
  unsigned short* kbm   = qb + (size_t)TT * AA;
  unsigned short* wvt   = kbm + (size_t)TT * AA;
  unsigned short* vtm   = wvt + (size_t)CC * CC;
  unsigned short* xembB = vtm + (size_t)CC * TT;

  wv_transpose_kernel<<<dim3((CC * CC) / 256), dim3(256), 0, stream>>>(Wv, wvt);
  xemb_cvt_kernel<<<dim3((TT * CC) / (256 * 8)), dim3(256), 0, stream>>>(xemb,
                                                                         xembB);
  ln_qk_kernel<<<dim3(TT), dim3(256), 0, stream>>>(x, sc, bi, Wq, Wk, qb, kbm);
  gemm_v_kernel<<<dim3(TT / 128, CC / 64), dim3(256), 0, stream>>>(xembB, wvt,
                                                                   vtm);
  attn_pv_kernel<<<dim3(TT / 128, CC / 128), dim3(256), 0, stream>>>(x, qb, kbm,
                                                                     vtm, out);
}
